// tgGAT_18150531793496
// MI455X (gfx1250) — matrix operand order repaired and checked by reading, not compiled, not run on hardware
//
#include <hip/hip_runtime.h>
#include <hip/hip_bf16.h>
#include <math.h>

// ---------------------------------------------------------------------------
// GAT 2-layer forward for MI455X (gfx1250, wave32, WMMA, LDS-staged GEMM).
// ---------------------------------------------------------------------------

typedef __attribute__((ext_vector_type(16))) __bf16 v16bf;
typedef __attribute__((ext_vector_type(8)))  float  v8f;

static constexpr long long NN = 100000;              // nodes
static constexpr long long EE = 1600000;             // directed edges
static constexpr long long ET = EE + NN;             // edges + self loops
static constexpr int D_IN = 128;
static constexpr int H    = 8;                       // heads (both layers)
static constexpr int F1   = 512;                     // H1*C1
static constexpr int F2   = 128;                     // H2*C2
static constexpr int C1c  = 64;
static constexpr int C2c  = 16;
static constexpr float NEG_SLOPE = 0.2f;
static constexpr float EPSV = 1e-16f;

// ---- workspace layout (float offsets) ----
static constexpr long long OFF_H    = 0;                       // h1 [N,512] / h2 [N,128] (reused)
static constexpr long long OFF_X1   = NN * F1;                 // x1 [N,512]
static constexpr long long OFF_AGG2 = 2 * NN * F1;             // agg2 [N,128]
static constexpr long long OFF_AS   = OFF_AGG2 + NN * F2;      // [N,8]
static constexpr long long OFF_AD   = OFF_AS + NN * H;         // [N,8]
static constexpr long long OFF_M    = OFF_AD + NN * H;         // [N,8]
static constexpr long long OFF_DEN  = OFF_M + NN * H;          // [N,8]
static constexpr long long OFF_EP   = OFF_DEN + NN * H;        // [ET,8]

__device__ __forceinline__ void atomicMaxF(float* addr, float val) {
    // Sign-split monotonic trick: valid mixed-sign float max via integer atomics.
    if (val >= 0.0f)
        atomicMax((int*)addr, __float_as_int(val));
    else
        atomicMin((unsigned int*)addr, __float_as_uint(val));
}

// ---------------------------------------------------------------------------
__global__ void tg_fill(float* __restrict__ p, long long n, float v) {
    long long i = blockIdx.x * (long long)blockDim.x + threadIdx.x;
    if (i < n) p[i] = v;
}

// ---------------------------------------------------------------------------
// C[M,Nd] = A[M,Kd] (fp32) x B[Kd,Nd] (fp32), bf16 WMMA, fp32 accumulate.
// Block = 256 threads = 8 waves. Each wave computes a 16x64 C tile (4 WMMAs
// per 32-deep k-step, one shared A fragment). B tile (32x64 fp32) is staged
// in LDS once per k-step with coalesced float4 loads shared by all 8 waves.
// grid.x = Nd/64, grid.y = ceil((M/16)/8). M % 16 == 0.
__global__ void __launch_bounds__(256)
tg_gemm_wmma(const float* __restrict__ A,
             const float* __restrict__ B,
             float* __restrict__ C,
             int M, int Kd, int Nd) {
    __shared__ float Bs[32][65];                     // pad to 65: bank-conflict free

    const int lane = threadIdx.x & 31;
    const int w    = threadIdx.x >> 5;
    const int l15  = lane & 15;
    const int half = lane >> 4;

    const int n0     = blockIdx.x * 64;
    const int tilesM = M >> 4;
    const int tmRaw  = blockIdx.y * 8 + w;
    const int tm     = tmRaw < tilesM ? tmRaw : tilesM - 1;  // clamp: EXEC stays all-1s
    const long long row = (long long)tm * 16 + l15;

    v8f acc[4] = {};
    for (int k0 = 0; k0 < Kd; k0 += 32) {
        // ---- stage B[k0..k0+31][n0..n0+63] into LDS (coalesced float4) ----
        __syncthreads();
        for (int f = threadIdx.x; f < 512; f += 256) {
            int r  = f >> 4;
            int c4 = (f & 15) << 2;
            float4 v = *(const float4*)(B + (long long)(k0 + r) * Nd + n0 + c4);
            Bs[r][c4 + 0] = v.x; Bs[r][c4 + 1] = v.y;
            Bs[r][c4 + 2] = v.z; Bs[r][c4 + 3] = v.w;
        }
        __syncthreads();

        // ---- A 16x32 bf16 fragment: lane holds K = k0+half*8+{0..7}, +16 ----
        v16bf a;
        const float* arow = A + row * (long long)Kd + k0 + half * 8;
        float4 a0 = *(const float4*)(arow + 0);
        float4 a1 = *(const float4*)(arow + 4);
        float4 a2 = *(const float4*)(arow + 16);
        float4 a3 = *(const float4*)(arow + 20);
        a[0]  = (__bf16)a0.x; a[1]  = (__bf16)a0.y; a[2]  = (__bf16)a0.z; a[3]  = (__bf16)a0.w;
        a[4]  = (__bf16)a1.x; a[5]  = (__bf16)a1.y; a[6]  = (__bf16)a1.z; a[7]  = (__bf16)a1.w;
        a[8]  = (__bf16)a2.x; a[9]  = (__bf16)a2.y; a[10] = (__bf16)a2.z; a[11] = (__bf16)a2.w;
        a[12] = (__bf16)a3.x; a[13] = (__bf16)a3.y; a[14] = (__bf16)a3.z; a[15] = (__bf16)a3.w;

        // ---- 4 B fragments from LDS, 4 WMMAs reusing the A fragment ----
#pragma unroll
        for (int j = 0; j < 4; ++j) {
            v16bf b;                                 // lane = column, K = half*8+i (+8 for i>=8)
#pragma unroll
            for (int i = 0; i < 16; ++i)
                b[i] = (__bf16)Bs[half * 8 + i + (i & 8)][j * 16 + l15];
            acc[j] = __builtin_amdgcn_wmma_f32_16x16x32_bf16(
                         false, a, false, b, (short)0, acc[j], false, false);
        }
    }

    if (tmRaw < tilesM) {
        // C layout: VGPR r -> row tm*16 + half*8 + r, col = n-tile + lane&15.
        float* crow = C + ((long long)tm * 16 + half * 8) * Nd + n0 + l15;
#pragma unroll
        for (int j = 0; j < 4; ++j)
#pragma unroll
            for (int r = 0; r < 8; ++r)
                crow[(long long)r * Nd + j * 16] = acc[j][r];
    }
}

// ---------------------------------------------------------------------------
// Attention score projections: as[n,h] = sum_c h[n,h,c]*a_src[h,c]; same for ad.
__global__ void tg_scores(const float* __restrict__ hbuf,
                          const float* __restrict__ a_src,
                          const float* __restrict__ a_dst,
                          float* __restrict__ as_, float* __restrict__ ad_,
                          int stride, int cph) {
    long long t = blockIdx.x * (long long)blockDim.x + threadIdx.x;
    if (t >= NN * H) return;
    long long n = t >> 3;
    int h = (int)(t & 7);
    const float* hp = hbuf + n * stride + h * cph;
    const float* sp = a_src + h * cph;
    const float* dp = a_dst + h * cph;
    float ss = 0.0f, sd = 0.0f;
    for (int c = 0; c < cph; ++c) {
        float v = hp[c];
        ss += v * sp[c];
        sd += v * dp[c];
    }
    as_[t] = ss;
    ad_[t] = sd;
}

__device__ __forceinline__ void edge_src_dst(const int* ei, long long e,
                                             long long& src, long long& dst) {
    if (e < EE) { src = ei[e]; dst = ei[EE + e]; }
    else        { src = e - EE; dst = e - EE; }      // self loops
}

// ---------------------------------------------------------------------------
__global__ void tg_edge_max(const int* __restrict__ ei,
                            const float* __restrict__ as_,
                            const float* __restrict__ ad_,
                            float* __restrict__ m) {
    long long e = blockIdx.x * (long long)blockDim.x + threadIdx.x;
    if (e >= ET) return;
    long long src, dst;
    edge_src_dst(ei, e, src, dst);
    const float4* s4 = (const float4*)(as_ + src * H);
    const float4* d4 = (const float4*)(ad_ + dst * H);
#pragma unroll
    for (int q = 0; q < 2; ++q) {
        float4 s = s4[q], d = d4[q];
        float v0 = s.x + d.x, v1 = s.y + d.y, v2 = s.z + d.z, v3 = s.w + d.w;
        v0 = v0 > 0.f ? v0 : NEG_SLOPE * v0;
        v1 = v1 > 0.f ? v1 : NEG_SLOPE * v1;
        v2 = v2 > 0.f ? v2 : NEG_SLOPE * v2;
        v3 = v3 > 0.f ? v3 : NEG_SLOPE * v3;
        float* mp = m + dst * H + q * 4;
        atomicMaxF(mp + 0, v0);
        atomicMaxF(mp + 1, v1);
        atomicMaxF(mp + 2, v2);
        atomicMaxF(mp + 3, v3);
    }
}

// ---------------------------------------------------------------------------
__global__ void tg_edge_expsum(const int* __restrict__ ei,
                               const float* __restrict__ as_,
                               const float* __restrict__ ad_,
                               const float* __restrict__ m,
                               float* __restrict__ den,
                               float* __restrict__ ep) {
    long long e = blockIdx.x * (long long)blockDim.x + threadIdx.x;
    if (e >= ET) return;
    long long src, dst;
    edge_src_dst(ei, e, src, dst);
#pragma unroll
    for (int h = 0; h < H; ++h) {
        float v = as_[src * H + h] + ad_[dst * H + h];
        v = v > 0.f ? v : NEG_SLOPE * v;
        float p = __expf(v - m[dst * H + h]);
        ep[e * H + h] = p;
        atomicAdd(den + dst * H + h, p);
    }
}

// ---------------------------------------------------------------------------
// out[dst, h*cph + c] += (ep/(den+eps)) * h[src, h*cph + c]
__global__ void tg_edge_agg(const int* __restrict__ ei,
                            const float* __restrict__ ep,
                            const float* __restrict__ den,
                            const float* __restrict__ hbuf,
                            float* __restrict__ outbuf,
                            int stride, int cph) {
    long long t = blockIdx.x * (long long)blockDim.x + threadIdx.x;
    if (t >= ET * H) return;
    long long e = t >> 3;
    int h = (int)(t & 7);
    long long src, dst;
    edge_src_dst(ei, e, src, dst);
    const float4* sp = (const float4*)(hbuf + src * stride + h * cph);
    __builtin_prefetch(sp, 0, 0);                    // global_prefetch_b8: random gather
    float alpha = ep[t] / (den[dst * H + h] + EPSV);
    float* dp = outbuf + dst * stride + h * cph;
    int n4 = cph >> 2;
    for (int q = 0; q < n4; ++q) {
        float4 v = sp[q];
        atomicAdd(dp + q * 4 + 0, alpha * v.x);
        atomicAdd(dp + q * 4 + 1, alpha * v.y);
        atomicAdd(dp + q * 4 + 2, alpha * v.z);
        atomicAdd(dp + q * 4 + 3, alpha * v.w);
    }
}

// ---------------------------------------------------------------------------
__global__ void tg_bias_add(float* __restrict__ x, const float* __restrict__ b,
                            long long total, int mod) {
    long long i = blockIdx.x * (long long)blockDim.x + threadIdx.x;
    if (i < total) x[i] += b[i % mod];
}

// ---------------------------------------------------------------------------
// mean over heads + b2, ELU, log_softmax over 16 classes.
__global__ void tg_finalize(const float* __restrict__ agg,
                            const float* __restrict__ b2,
                            float* __restrict__ out) {
    long long n = blockIdx.x * (long long)blockDim.x + threadIdx.x;
    if (n >= NN) return;
    float acc[C2c];
#pragma unroll
    for (int c = 0; c < C2c; ++c) acc[c] = 0.0f;
    const float* p = agg + n * F2;
#pragma unroll
    for (int h = 0; h < H; ++h)
#pragma unroll
        for (int c = 0; c < C2c; ++c)
            acc[c] += p[h * C2c + c];
    float mx = -INFINITY;
#pragma unroll
    for (int c = 0; c < C2c; ++c) {
        float y = acc[c] * 0.125f + b2[c];
        y = y > 0.f ? y : (__expf(y) - 1.0f);        // ELU (alpha=1)
        acc[c] = y;
        mx = fmaxf(mx, y);
    }
    float s = 0.0f;
#pragma unroll
    for (int c = 0; c < C2c; ++c) s += __expf(acc[c] - mx);
    float lse = mx + __logf(s);
#pragma unroll
    for (int c = 0; c < C2c; ++c)
        out[n * C2c + c] = acc[c] - lse;
}

// ---------------------------------------------------------------------------
static inline int nblk(long long n, int b) { return (int)((n + b - 1) / b); }

extern "C" void kernel_launch(void* const* d_in, const int* in_sizes, int n_in,
                              void* d_out, int out_size, void* d_ws, size_t ws_size,
                              hipStream_t stream) {
    const float* x      = (const float*)d_in[0];
    const int*   ei     = (const int*)  d_in[1];
    const float* W1     = (const float*)d_in[2];   // [128,512] row-major
    const float* a1s    = (const float*)d_in[3];
    const float* a1d    = (const float*)d_in[4];
    const float* b1     = (const float*)d_in[5];   // [512]
    const float* W2     = (const float*)d_in[6];   // [512,128] row-major
    const float* a2s    = (const float*)d_in[7];
    const float* a2d    = (const float*)d_in[8];
    const float* b2     = (const float*)d_in[9];   // [16]
    float* out = (float*)d_out;

    float* ws   = (float*)d_ws;
    float* h12  = ws + OFF_H;        // h1 [N,512], later h2 [N,128]
    float* x1   = ws + OFF_X1;
    float* agg2 = ws + OFF_AGG2;
    float* as_  = ws + OFF_AS;
    float* ad_  = ws + OFF_AD;
    float* m    = ws + OFF_M;
    float* den  = ws + OFF_DEN;
    float* ep   = ws + OFF_EP;

    const int B = 256;
    const int tilesM  = (int)(NN / 16);              // 6250
    const int gridYmm = (tilesM + 7) / 8;            // 782

    // ---------------- layer 1 ----------------
    tg_fill<<<nblk(NN * F1, B), B, 0, stream>>>(x1, NN * F1, 0.0f);
    tg_fill<<<nblk(NN * H, B), B, 0, stream>>>(m, NN * H, -INFINITY);
    tg_fill<<<nblk(NN * H, B), B, 0, stream>>>(den, NN * H, 0.0f);

    {   // h1 = x @ W1   (100000x128 x 128x512)
        dim3 grid(F1 / 64, gridYmm);
        tg_gemm_wmma<<<grid, B, 0, stream>>>(x, W1, h12, (int)NN, D_IN, F1);
    }
    tg_scores<<<nblk(NN * H, B), B, 0, stream>>>(h12, a1s, a1d, as_, ad_, F1, C1c);
    tg_edge_max<<<nblk(ET, B), B, 0, stream>>>(ei, as_, ad_, m);
    tg_edge_expsum<<<nblk(ET, B), B, 0, stream>>>(ei, as_, ad_, m, den, ep);
    tg_edge_agg<<<nblk(ET * H, B), B, 0, stream>>>(ei, ep, den, h12, x1, F1, C1c);
    tg_bias_add<<<nblk(NN * F1, B), B, 0, stream>>>(x1, b1, NN * F1, F1);

    // ---------------- layer 2 ----------------
    tg_fill<<<nblk(NN * F2, B), B, 0, stream>>>(agg2, NN * F2, 0.0f);
    tg_fill<<<nblk(NN * H, B), B, 0, stream>>>(m, NN * H, -INFINITY);
    tg_fill<<<nblk(NN * H, B), B, 0, stream>>>(den, NN * H, 0.0f);

    {   // h2 = x1 @ W2  (100000x512 x 512x128), reuses h1's region
        dim3 grid(F2 / 64, gridYmm);
        tg_gemm_wmma<<<grid, B, 0, stream>>>(x1, W2, h12, (int)NN, F1, F2);
    }
    tg_scores<<<nblk(NN * H, B), B, 0, stream>>>(h12, a2s, a2d, as_, ad_, F2, C2c);
    tg_edge_max<<<nblk(ET, B), B, 0, stream>>>(ei, as_, ad_, m);
    tg_edge_expsum<<<nblk(ET, B), B, 0, stream>>>(ei, as_, ad_, m, den, ep);
    tg_edge_agg<<<nblk(ET * H, B), B, 0, stream>>>(ei, ep, den, h12, agg2, F2, C2c);

    tg_finalize<<<nblk(NN, B), B, 0, stream>>>(agg2, b2, out);
}
